// SS2DBlock_5617817224161
// MI455X (gfx1250) — compile-verified, matrix operand order checked by reading, not run on hardware
//
#include <hip/hip_runtime.h>
#include <hip/hip_bf16.h>
#include <math.h>

typedef __attribute__((ext_vector_type(16))) __bf16 v16bf;
typedef __attribute__((ext_vector_type(8)))  float  v8f;
typedef __attribute__((ext_vector_type(4)))  unsigned int u32x4;
typedef __attribute__((ext_vector_type(4)))  int i32x4;
typedef __attribute__((ext_vector_type(8)))  int i32x8;

#define L_IMG   4096
#define M_TOK   65536     // B * H * W
#define DINNER  256

// ---------------- fragment loaders (16-bit A/B layout per CDNA5 ISA 7.12.2) ---

template<int KD>
__device__ inline v16bf frag_ld(const __bf16* row, int k0, int hi) {
  union { uint4 u[2]; v16bf v; } f;
  if (KD % 32 == 0) {
    f.u[0] = *(const uint4*)(row + k0 + hi * 8);        // K in [k0+hi*8, +8)
    f.u[1] = *(const uint4*)(row + k0 + 16 + hi * 8);   // K in [k0+16+hi*8, +8)
  } else {                                              // KD == 16, k0 == 0
    f.u[0] = *(const uint4*)(row + hi * 8);             // K in [hi*8, +8)
    f.u[1] = make_uint4(0u, 0u, 0u, 0u);                // K >= 16 padded
  }
  return f.v;
}

// ---------------- generic bf16 WMMA GEMM, fused epilogues ---------------------
// C[M x NN] = A[M x KD] (tokens, bf16) * W[NN x KD]^T (weights, bf16)
// block = 128 threads = 4 waves in 2x2; wave tile 32x32 (4 WMMA accumulators)

struct GemmP {
  const __bf16* A; const __bf16* W; const float* bias;
  float* out0; float* out1; __bf16* outb; const float* extra;
};

template<int EPI, int KD, int NN, int REV>
__global__ __launch_bounds__(128) void gemm_wmma(GemmP p) {
  static_assert(NN % 64 == 0, "NN padded to 64");
  const int tid = threadIdx.x, wave = tid >> 5, lane = tid & 31;
  const int hi = lane >> 4, lr = lane & 15;
  const int m0 = blockIdx.x * 64 + (wave >> 1) * 32;
  const int n0 = blockIdx.y * 64 + (wave & 1) * 32;

  const v8f zacc = {0.f, 0.f, 0.f, 0.f, 0.f, 0.f, 0.f, 0.f};
  v8f acc[2][2] = {{zacc, zacc}, {zacc, zacc}};

  const __bf16* arow[2];
#pragma unroll
  for (int mi = 0; mi < 2; ++mi) {
    int r = m0 + mi * 16 + lr;
    if (REV) r = (r & ~(L_IMG - 1)) | ((L_IMG - 1) - (r & (L_IMG - 1)));
    arow[mi] = p.A + (size_t)r * KD;
  }
  const __bf16* wrow[2];
#pragma unroll
  for (int ni = 0; ni < 2; ++ni)
    wrow[ni] = p.W + (size_t)(n0 + ni * 16 + lr) * KD;

#pragma unroll 2
  for (int k0 = 0; k0 < KD; k0 += 32) {
    if (k0 + 32 < KD) {               // global_prefetch_b8, near/high-temporal
      __builtin_prefetch(arow[0] + k0 + 32, 0, 3);
      __builtin_prefetch(arow[1] + k0 + 32, 0, 3);
    }
    v16bf a0 = frag_ld<KD>(arow[0], k0, hi);
    v16bf a1 = frag_ld<KD>(arow[1], k0, hi);
    v16bf b0 = frag_ld<KD>(wrow[0], k0, hi);
    v16bf b1 = frag_ld<KD>(wrow[1], k0, hi);
    acc[0][0] = __builtin_amdgcn_wmma_f32_16x16x32_bf16(false, a0, false, b0, (short)0, acc[0][0], false, false);
    acc[0][1] = __builtin_amdgcn_wmma_f32_16x16x32_bf16(false, a0, false, b1, (short)0, acc[0][1], false, false);
    acc[1][0] = __builtin_amdgcn_wmma_f32_16x16x32_bf16(false, a1, false, b0, (short)0, acc[1][0], false, false);
    acc[1][1] = __builtin_amdgcn_wmma_f32_16x16x32_bf16(false, a1, false, b1, (short)0, acc[1][1], false, false);
  }

#pragma unroll
  for (int mi = 0; mi < 2; ++mi)
#pragma unroll
  for (int ni = 0; ni < 2; ++ni) {
    const int cn = n0 + ni * 16 + lr;
#pragma unroll
    for (int r = 0; r < 8; ++r) {
      const int row = m0 + mi * 16 + hi * 8 + r;
      float a = acc[mi][ni][r];
      if (EPI == 0) {                 // bias + store f32 (ld = NN)
        p.out0[(size_t)row * NN + cn] = a + p.bias[cn];
      } else if (EPI == 1) {          // in_proj: split xi(silu,f32+bf16) / z(f32)
        float t = a + p.bias[cn];
        if (cn < 256) {
          float s = t / (1.f + __expf(-t));
          p.out0[(size_t)row * 256 + cn] = s;
          p.outb[(size_t)row * 256 + cn] = (__bf16)s;
        } else {
          p.out1[(size_t)row * 256 + (cn - 256)] = t;
        }
      } else if (EPI == 2) {          // x_proj (no bias): dt->bf16, B/C->f32
        if (cn < 16)       p.outb[(size_t)row * 16 + cn] = (__bf16)a;
        else if (cn < 48)  p.out0[(size_t)row * 32 + (cn - 16)] = a;
      } else if (EPI == 3) {          // dt_proj: bias + softplus, f32 (ld 256)
        float t = a + p.bias[cn];
        float sp = (t > 20.f) ? t : log1pf(__expf(t));
        p.out0[(size_t)row * 256 + cn] = sp;
      } else if (EPI == 5) {          // fc2: bias + residual, store channels-first
        int b = row >> 12, l = row & (L_IMG - 1);
        size_t o = ((size_t)(b * 128 + cn)) * L_IMG + l;
        p.out0[o] = a + p.bias[cn] + p.extra[o];
      }
    }
  }
}

// ---------------- small conversion kernels ------------------------------------

__global__ __launch_bounds__(256) void cvt_f32_bf16(const float* s, __bf16* d, int n) {
  int i = blockIdx.x * 256 + threadIdx.x;
  if (i < n) d[i] = (__bf16)s[i];
}

__global__ __launch_bounds__(256) void cvt_pad_rows(const float* s, __bf16* d,
                                                    int rows, int rowsPad, int cols) {
  int i = blockIdx.x * 256 + threadIdx.x;
  if (i >= rowsPad * cols) return;
  int r = i / cols, c = i - r * cols;
  d[i] = (r < rows) ? (__bf16)s[r * cols + c] : (__bf16)0.0f;
}

// x (B,128,HW) channels-first f32 -> Xt (B*HW, 128) token-major bf16
__global__ __launch_bounds__(256) void cvt_x_tokens(const float* x, __bf16* Xt) {
  size_t i = (size_t)blockIdx.x * 256 + threadIdx.x;
  if (i >= (size_t)M_TOK * 128) return;
  int c = (int)(i & 127);
  size_t t = i >> 7;
  int b = (int)(t >> 12), l = (int)(t & (L_IMG - 1));
  Xt[i] = (__bf16)x[((size_t)b * 128 + c) * L_IMG + l];
}

// ---------------- depthwise 3x3 + bias + SiLU (token-major) -------------------

__global__ __launch_bounds__(256) void dwconv_silu(const float* __restrict__ H1,
                                                   const float* __restrict__ w,
                                                   const float* __restrict__ bias,
                                                   __bf16* __restrict__ out) {
  const int c = threadIdx.x;            // channel 0..255
  const int m = blockIdx.x;             // token
  const int b = m >> 12, l = m & (L_IMG - 1);
  const int h = l >> 6, wx = l & 63;
  float acc = bias[c];
#pragma unroll
  for (int dy = -1; dy <= 1; ++dy)
#pragma unroll
    for (int dx = -1; dx <= 1; ++dx) {
      int hh = h + dy, ww = wx + dx;
      if (hh >= 0 && hh < 64 && ww >= 0 && ww < 64) {
        int mm = (b << 12) | (hh << 6) | ww;
        acc = fmaf(w[c * 9 + (dy + 1) * 3 + (dx + 1)], H1[(size_t)mm * 256 + c], acc);
      }
    }
  float s = acc / (1.f + __expf(-acc));
  out[(size_t)m * 256 + c] = (__bf16)s;
}

// ---------------- TDM: 1-D tile DMA global->LDS (Tensor Data Mover) -----------
// Packs a D# for a contiguous run of `nelem` DWORDs (data_size=4B), 1-D tile.
// group0: count=1, lds_addr, global_addr[56:0], type=2 ("image")
// group1: mask=0, data_size=2(4B), tensor_dim0=tile_dim0=nelem, dims 1/2 unused

__device__ inline void tdm_load_1d(void* lds_dst, const void* gsrc, unsigned nelem) {
  unsigned long long ga = (unsigned long long)(uintptr_t)gsrc;
  unsigned lds_off = (unsigned)(uintptr_t)lds_dst;     // low 32 bits = LDS offset
  u32x4 g0;
  g0.x = 0x1u;                                         // count=1, user descriptor
  g0.y = lds_off;                                      // lds_addr (bytes)
  g0.z = (unsigned)(ga & 0xffffffffu);                 // global_addr[31:0]
  g0.w = (unsigned)((ga >> 32) & 0x01ffffffu) | (2u << 30);  // addr[56:32] | type=2
  i32x8 g1;
  g1[0] = 0x00020000;                                  // data_size=2 (4 bytes)
  g1[1] = (int)((nelem & 0xffffu) << 16);              // tensor_dim0[15:0]
  g1[2] = (int)(nelem >> 16);                          // tensor_dim0[31:16] | dim1=0
  g1[3] = (int)((nelem & 0xffffu) << 16);              // tile_dim0 = nelem
  g1[4] = 0;                                           // tile_dim1/2 unused
  g1[5] = (int)nelem;                                  // tensor_dim0_stride[31:0]
  g1[6] = 0;
  g1[7] = 0;
  i32x4 z4 = {0, 0, 0, 0};
#if defined(__clang_major__) && (__clang_major__ >= 23)
  i32x8 z8 = {0, 0, 0, 0, 0, 0, 0, 0};
  __builtin_amdgcn_tensor_load_to_lds(g0, g1, z4, z4, z8, 0);
#else
  __builtin_amdgcn_tensor_load_to_lds(g0, g1, z4, z4, 0);
#endif
}

// ---------------- selective scan ----------------------------------------------
// One block = 64 channels of one (b,k); each lane owns 16 states in regs.
// B/C chunks (32 steps x 32 f32 = 4KB, contiguous) are DMA'd global->LDS by the
// Tensor Data Mover, double-buffered so the DMA overlaps the exp/fma chain.

__global__ __launch_bounds__(64) void scan_kernel(const float* __restrict__ xi,
                                                  const float* __restrict__ dtsp,
                                                  const float* __restrict__ BC,
                                                  const float* __restrict__ A_log,
                                                  const float* __restrict__ Ds,
                                                  float* __restrict__ ys) {
  const int gid = blockIdx.x;           // B*K*4 = 128 blocks
  const int cblk = gid & 3;
  const int k = (gid >> 2) & 1;
  const int b = gid >> 3;
  const int c = cblk * 64 + threadIdx.x;

  const float* dts = dtsp + (size_t)k * M_TOK * 256;
  const float* bc  = BC   + (size_t)k * M_TOK * 32;
  float*       yo  = ys   + (size_t)k * M_TOK * 256;

  float An[16];
#pragma unroll
  for (int n = 0; n < 16; ++n) An[n] = -__expf(A_log[((size_t)(k * 256 + c)) * 16 + n]);
  const float Dv = Ds[k * 256 + c];

  float hst[16];
#pragma unroll
  for (int n = 0; n < 16; ++n) hst[n] = 0.f;

  __shared__ float sBC[2][32][32];      // double-buffered [chunk][step][B|C]

  // prologue: DMA chunk 0 into buffer 0
  tdm_load_1d(&sBC[0][0][0], bc + (size_t)b * L_IMG * 32, 1024);
  __builtin_amdgcn_s_wait_tensorcnt(0);
  __syncthreads();

  for (int l0 = 0; l0 < L_IMG; l0 += 32) {
    const int cur = (l0 >> 5) & 1;
    const bool more = (l0 + 32) < L_IMG;
    if (more)                           // overlap next chunk's DMA with compute
      tdm_load_1d(&sBC[cur ^ 1][0][0],
                  bc + ((size_t)b * L_IMG + l0 + 32) * 32, 1024);
#pragma unroll 4
    for (int li = 0; li < 32; ++li) {
      size_t m = (size_t)b * L_IMG + l0 + li;
      float dt = dts[m * 256 + c];
      size_t tokx = (k == 0) ? m : ((size_t)b * L_IMG + (L_IMG - 1 - (l0 + li)));
      float xv = xi[tokx * 256 + c];
      float du = dt * xv;
      float y = 0.f;
#pragma unroll
      for (int n = 0; n < 16; ++n) {
        float dA = __expf(dt * An[n]);
        hst[n] = fmaf(hst[n], dA, du * sBC[cur][li][n]);
        y = fmaf(hst[n], sBC[cur][li][16 + n], y);
      }
      yo[m * 256 + c] = fmaf(Dv, xv, y);
    }
    if (more) {
      __builtin_amdgcn_s_wait_tensorcnt(0);   // own DMA done
      __syncthreads();                        // all waves' DMA done, reads done
    }
  }
}

// ---------------- LayerNorm over 256 channels (+ optional dir-merge & z-gate) -

__global__ __launch_bounds__(256) void ln_gate(const float* __restrict__ y0,
                                               const float* __restrict__ y1,
                                               const float* __restrict__ z,
                                               const float* __restrict__ w,
                                               const float* __restrict__ bias,
                                               __bf16* __restrict__ out,
                                               int merge) {
  const int m = blockIdx.x, c = threadIdx.x;
  float y;
  if (merge) {
    int b = m >> 12, l = m & (L_IMG - 1);
    size_t mr = ((size_t)(b << 12) + (L_IMG - 1 - l)) * 256;
    y = y0[(size_t)m * 256 + c] + y1[mr + c];
  } else {
    y = y0[(size_t)m * 256 + c];
  }
  __shared__ float red[256];
  red[c] = y; __syncthreads();
#pragma unroll
  for (int s = 128; s > 0; s >>= 1) { if (c < s) red[c] += red[c + s]; __syncthreads(); }
  float mean = red[0] * (1.f / 256.f);
  __syncthreads();
  float d = y - mean;
  red[c] = d * d; __syncthreads();
#pragma unroll
  for (int s = 128; s > 0; s >>= 1) { if (c < s) red[c] += red[c + s]; __syncthreads(); }
  float var = red[0] * (1.f / 256.f);
  float v = d * rsqrtf(var + 1e-6f) * w[c] + bias[c];
  if (z) {
    float g = z[(size_t)m * 256 + c];
    v *= g / (1.f + __expf(-g));
  }
  out[(size_t)m * 256 + c] = (__bf16)v;
}

// ---------------- host orchestration ------------------------------------------

extern "C" void kernel_launch(void* const* d_in, const int* in_sizes, int n_in,
                              void* d_out, int out_size, void* d_ws, size_t ws_size,
                              hipStream_t stream) {
  (void)in_sizes; (void)n_in; (void)out_size; (void)ws_size;
  const float* x         = (const float*)d_in[0];
  const float* fc1_w     = (const float*)d_in[1];
  const float* fc1_b     = (const float*)d_in[2];
  const float* dw_w      = (const float*)d_in[3];
  const float* dw_b      = (const float*)d_in[4];
  const float* in_proj_w = (const float*)d_in[5];
  const float* in_proj_b = (const float*)d_in[6];
  const float* x_proj_w  = (const float*)d_in[7];
  const float* dt_proj_w = (const float*)d_in[8];
  const float* dt_proj_b = (const float*)d_in[9];
  const float* A_log     = (const float*)d_in[10];
  const float* Ds        = (const float*)d_in[11];
  const float* onw       = (const float*)d_in[12];
  const float* onb       = (const float*)d_in[13];
  const float* out_proj_w= (const float*)d_in[14];
  const float* out_proj_b= (const float*)d_in[15];
  const float* norm_w    = (const float*)d_in[16];
  const float* norm_b    = (const float*)d_in[17];
  const float* fc2_w     = (const float*)d_in[18];
  const float* fc2_b     = (const float*)d_in[19];

  char* wsb = (char*)d_ws; size_t off = 0;
  auto arena = [&](size_t bytes) -> char* {
    off = (off + 255) & ~(size_t)255;
    char* p = wsb + off; off += bytes; return p;
  };
  const size_t M = M_TOK;
  __bf16* Xt   = (__bf16*)arena(M * 128 * 2);
  float*  H1   = (float*) arena(M * 256 * 4);
  __bf16* H2b  = (__bf16*)arena(M * 256 * 2);
  float*  xif  = (float*) arena(M * 256 * 4);
  __bf16* xib  = (__bf16*)arena(M * 256 * 2);
  float*  zf   = (float*) arena(M * 256 * 4);
  __bf16* dtr  = (__bf16*)arena(2 * M * 16 * 2);
  float*  BCf  = (float*) arena(2 * M * 32 * 4);
  float*  dtsp = (float*) arena((size_t)2 * M * 256 * 4);
  float*  ysb  = (float*) arena((size_t)2 * M * 256 * 4);
  __bf16* wfc1 = (__bf16*)arena(256 * 128 * 2);
  __bf16* winp = (__bf16*)arena(512 * 256 * 2);
  __bf16* wxp  = (__bf16*)arena(2 * 64 * 256 * 2);   // 48 rows padded to 64 (zeros)
  __bf16* wdtp = (__bf16*)arena(2 * 256 * 16 * 2);
  __bf16* wop  = (__bf16*)arena(256 * 256 * 2);
  __bf16* wfc2 = (__bf16*)arena(128 * 256 * 2);
  // dead-buffer reuse
  float*  yof  = H1;                    // H1 dead after dwconv
  __bf16* ynb  = H2b;                   // H2b dead after in_proj GEMM
  __bf16* yob  = xib;                   // xib dead after x_proj GEMMs

  auto cvt = [&](const float* s, __bf16* d, int n) {
    cvt_f32_bf16<<<(n + 255) / 256, 256, 0, stream>>>(s, d, n);
  };
  cvt(fc1_w, wfc1, 256 * 128);
  cvt(in_proj_w, winp, 512 * 256);
  cvt(dt_proj_w, wdtp, 2 * 256 * 16);
  cvt(out_proj_w, wop, 256 * 256);
  cvt(fc2_w, wfc2, 128 * 256);
  for (int k = 0; k < 2; ++k)
    cvt_pad_rows<<<(64 * 256 + 255) / 256, 256, 0, stream>>>(
        x_proj_w + (size_t)k * 48 * 256, wxp + (size_t)k * 64 * 256, 48, 64, 256);

  cvt_x_tokens<<<(int)((M * 128 + 255) / 256), 256, 0, stream>>>(x, Xt);

  GemmP g;
  // fc1: (M,128)x(256,128) -> H1 f32
  g = {Xt, wfc1, fc1_b, H1, nullptr, nullptr, nullptr};
  gemm_wmma<0, 128, 256, 0><<<dim3(M / 64, 4), 128, 0, stream>>>(g);

  dwconv_silu<<<(int)M, 256, 0, stream>>>(H1, dw_w, dw_b, H2b);

  // in_proj: (M,256)x(512,256) -> silu(xi) f32+bf16, z f32
  g = {H2b, winp, in_proj_b, xif, zf, xib, nullptr};
  gemm_wmma<1, 256, 512, 0><<<dim3(M / 64, 8), 128, 0, stream>>>(g);

  // x_proj per direction: (M,256)x(64pad,256) -> dt bf16, B/C f32 (k=1 reversed)
  g = {xib, wxp, nullptr, BCf, nullptr, dtr, nullptr};
  gemm_wmma<2, 256, 64, 0><<<dim3(M / 64, 1), 128, 0, stream>>>(g);
  g = {xib, wxp + (size_t)64 * 256, nullptr, BCf + M * 32, nullptr, dtr + M * 16, nullptr};
  gemm_wmma<2, 256, 64, 1><<<dim3(M / 64, 1), 128, 0, stream>>>(g);

  // dt_proj per direction: (M,16)x(256,16) + bias, softplus -> f32
  for (int k = 0; k < 2; ++k) {
    g = {dtr + (size_t)k * M * 16, wdtp + (size_t)k * 256 * 16, dt_proj_b + k * 256,
         dtsp + (size_t)k * M * 256, nullptr, nullptr, nullptr};
    gemm_wmma<3, 16, 256, 0><<<dim3(M / 64, 4), 128, 0, stream>>>(g);
  }

  // selective scan (sequential over L, parallel over B*K*C; TDM-fed LDS)
  scan_kernel<<<128, 64, 0, stream>>>(xif, dtsp, BCf, A_log, Ds, ysb);

  // out_norm (merge fwd+bwd) * silu(z) -> bf16
  ln_gate<<<(int)M, 256, 0, stream>>>(ysb, ysb + M * 256, zf, onw, onb, ynb, 1);

  // out_proj: (M,256)x(256,256) -> yo f32
  g = {ynb, wop, out_proj_b, yof, nullptr, nullptr, nullptr};
  gemm_wmma<0, 256, 256, 0><<<dim3(M / 64, 4), 128, 0, stream>>>(g);

  // final LayerNorm -> bf16
  ln_gate<<<(int)M, 256, 0, stream>>>(yof, nullptr, nullptr, norm_w, norm_b, yob, 0);

  // fc2 + residual -> d_out (B,128,64,64) channels-first f32
  g = {yob, wfc2, fc2_b, (float*)d_out, nullptr, nullptr, x};
  gemm_wmma<5, 256, 128, 0><<<dim3(M / 64, 2), 128, 0, stream>>>(g);
}
